// VectorContextLayer_45509473468509
// MI455X (gfx1250) — compile-verified, matrix operand order checked by reading, not run on hardware
//
#include <hip/hip_runtime.h>
#include <hip/hip_bf16.h>
#include <math.h>
#include <stdint.h>

// ---------------------------------------------------------------------------
// VectorContextLayer for MI455X (gfx1250): bf16 WMMA pipeline.
//   B=2, S=512, D=1024, H=16, HD=64, N=4096, VD=768, INJ=0.3
// Compute bound (~45 GFLOP vs ~55MB traffic) -> everything on
// v_wmma_f32_16x16x32_bf16; flash-style attention (never materialize scores);
// K/V tiles staged via global_load_async_to_lds_b128 (ASYNCcnt, double-buffer)
// and shared by all 4 waves of a block.
// ---------------------------------------------------------------------------

typedef __bf16 bf16_t;
typedef __attribute__((ext_vector_type(16))) __bf16 v16bf;
typedef __attribute__((ext_vector_type(8)))  float  v8f;
typedef __attribute__((ext_vector_type(4)))  float  v4f;

constexpr int Bc  = 2;
constexpr int Sc  = 512;
constexpr int Dc  = 1024;
constexpr int Hc  = 16;
constexpr int HDc = 64;
constexpr int Nctx = 4096;
constexpr int VDc = 768;
constexpr int BSc = Bc * Sc;          // 1024 rows
constexpr float INJf = 0.3f;

__device__ __forceinline__ v8f zero_v8f() {
  v8f z;
  #pragma unroll
  for (int i = 0; i < 8; ++i) z[i] = 0.0f;
  return z;
}

union V16U { v16bf v; v4f q[2]; };

// Load a 16x32 bf16 fragment in the WMMA A/B register layout from a
// row-major matrix (row stride `ld` elements). Lane l (l<16) supplies
// row l with K slots {0..7,16..23}; lane l+16 supplies row l with
// K slots {8..15,24..31}. Two b128 loads per lane (global or ds).
__device__ __forceinline__ v16bf load_frag(const bf16_t* base, int ld, int lane) {
  const int hf = (lane >> 4) & 1;
  const int r  = lane & 15;
  const bf16_t* p0 = base + (size_t)r * ld + hf * 8;
  V16U u;
  u.q[0] = *(const v4f*)(p0);
  u.q[1] = *(const v4f*)(p0 + 16);
  return u.v;
}

__device__ __forceinline__ v8f wmma_bf16(v16bf a, v16bf b, v8f c) {
  // (neg_a, A, neg_b, B, c_mod, C, reuse_a, reuse_b)
  return __builtin_amdgcn_wmma_f32_16x16x32_bf16(false, a, false, b,
                                                 (short)0, c, false, false);
}

// Async copy of one 16-byte chunk from global memory into LDS (ASYNCcnt).
__device__ __forceinline__ void async_b128(const bf16_t* gsrc, bf16_t* ldst) {
  unsigned laddr = (unsigned)(uintptr_t)ldst;             // LDS byte address
  unsigned long long gaddr = (unsigned long long)gsrc;
  asm volatile("global_load_async_to_lds_b128 %0, %1, off"
               :: "v"(laddr), "v"(gaddr) : "memory");
}

// ---------------------------------------------------------------------------
// Elementwise helpers
// ---------------------------------------------------------------------------
__global__ void f32_to_bf16_kernel(const float* __restrict__ src,
                                   bf16_t* __restrict__ dst, int n) {
  int i = blockIdx.x * 256 + threadIdx.x;
  if (i < n) dst[i] = (bf16_t)src[i];
}

// v: (B,S,H,HD) -> vt: (B,H,HD,S)
__global__ void transpose_v_kernel(const bf16_t* __restrict__ v,
                                   bf16_t* __restrict__ vt) {
  int idx = blockIdx.x * 256 + threadIdx.x;
  if (idx >= BSc * Dc) return;
  int d = idx % HDc;
  int h = (idx / HDc) % Hc;
  int s = (idx / (Hc * HDc)) % Sc;
  int b = idx / (Sc * Hc * HDc);
  vt[(((size_t)b * Hc + h) * HDc + d) * Sc + s] = v[idx];
}

// vv: (N, D) -> vvt: (D, N)
__global__ void transpose_nm_kernel(const bf16_t* __restrict__ src,
                                    bf16_t* __restrict__ dst) {
  int idx = blockIdx.x * 256 + threadIdx.x;
  if (idx >= Nctx * Dc) return;
  int c = idx % Dc;
  int r = idx / Dc;
  dst[(size_t)c * Nctx + r] = src[idx];
}

// ---------------------------------------------------------------------------
// C_bf16(MxN) = A_bf16(MxK) @ B_bf16(NxK)^T + bias_f32(N)
// One 32x32 C tile per wave; 4 WMMAs per 32-wide K step.
// ---------------------------------------------------------------------------
__global__ void gemm_bt_bias_bf16out(const bf16_t* __restrict__ A,
                                     const bf16_t* __restrict__ Bw,
                                     const float*  __restrict__ bias,
                                     bf16_t* __restrict__ C,
                                     int M, int Nn, int K, int totalTiles) {
  const int wave = threadIdx.x >> 5;
  const int lane = threadIdx.x & 31;
  const int t = blockIdx.x * 4 + wave;
  if (t >= totalTiles) return;                 // wave-uniform: EXEC stays full
  const int tilesN = Nn >> 5;
  const int tm = t / tilesN, tn = t % tilesN;
  const bf16_t* A0 = A  + (size_t)tm * 32 * K;
  const bf16_t* B0 = Bw + (size_t)tn * 32 * K;

  v8f acc00 = zero_v8f(), acc01 = zero_v8f();
  v8f acc10 = zero_v8f(), acc11 = zero_v8f();

  for (int k0 = 0; k0 < K; k0 += 32) {
    if (k0 + 32 < K) {
      __builtin_prefetch((const void*)(A0 + k0 + 32), 0, 1);
      __builtin_prefetch((const void*)(B0 + k0 + 32), 0, 1);
    }
    v16bf a0 = load_frag(A0 + k0, K, lane);
    v16bf a1 = load_frag(A0 + (size_t)16 * K + k0, K, lane);
    v16bf b0 = load_frag(B0 + k0, K, lane);
    v16bf b1 = load_frag(B0 + (size_t)16 * K + k0, K, lane);
    acc00 = wmma_bf16(a0, b0, acc00);
    acc01 = wmma_bf16(a0, b1, acc01);
    acc10 = wmma_bf16(a1, b0, acc10);
    acc11 = wmma_bf16(a1, b1, acc11);
  }

  const int hf = lane >> 4, n = lane & 15;
  const int col0 = tn * 32 + n, col1 = col0 + 16;
  const float bv0 = bias[col0], bv1 = bias[col1];
  #pragma unroll
  for (int v = 0; v < 8; ++v) {
    int r0 = tm * 32 + v + 8 * hf;
    int r1 = r0 + 16;
    C[(size_t)r0 * Nn + col0] = (bf16_t)(acc00[v] + bv0);
    C[(size_t)r0 * Nn + col1] = (bf16_t)(acc01[v] + bv1);
    C[(size_t)r1 * Nn + col0] = (bf16_t)(acc10[v] + bv0);
    C[(size_t)r1 * Nn + col1] = (bf16_t)(acc11[v] + bv1);
  }
}

// ---------------------------------------------------------------------------
// out_f32 = (1-INJ) * (A @ B^T + bias) + INJ * ctx      (out projection fuse)
// ---------------------------------------------------------------------------
__global__ void gemm_final_combine(const bf16_t* __restrict__ A,
                                   const bf16_t* __restrict__ Bw,
                                   const float*  __restrict__ bias,
                                   const float*  __restrict__ ctx,
                                   float* __restrict__ Out,
                                   int M, int Nn, int K, int totalTiles) {
  const int wave = threadIdx.x >> 5;
  const int lane = threadIdx.x & 31;
  const int t = blockIdx.x * 4 + wave;
  if (t >= totalTiles) return;
  const int tilesN = Nn >> 5;
  const int tm = t / tilesN, tn = t % tilesN;
  const bf16_t* A0 = A  + (size_t)tm * 32 * K;
  const bf16_t* B0 = Bw + (size_t)tn * 32 * K;

  v8f acc00 = zero_v8f(), acc01 = zero_v8f();
  v8f acc10 = zero_v8f(), acc11 = zero_v8f();

  for (int k0 = 0; k0 < K; k0 += 32) {
    v16bf a0 = load_frag(A0 + k0, K, lane);
    v16bf a1 = load_frag(A0 + (size_t)16 * K + k0, K, lane);
    v16bf b0 = load_frag(B0 + k0, K, lane);
    v16bf b1 = load_frag(B0 + (size_t)16 * K + k0, K, lane);
    acc00 = wmma_bf16(a0, b0, acc00);
    acc01 = wmma_bf16(a0, b1, acc01);
    acc10 = wmma_bf16(a1, b0, acc10);
    acc11 = wmma_bf16(a1, b1, acc11);
  }

  const int hf = lane >> 4, n = lane & 15;
  const int col0 = tn * 32 + n, col1 = col0 + 16;
  const float bv0 = bias[col0], bv1 = bias[col1];
  const float w0 = 1.0f - INJf, w1 = INJf;
  #pragma unroll
  for (int v = 0; v < 8; ++v) {
    int r0 = tm * 32 + v + 8 * hf;
    int r1 = r0 + 16;
    size_t i00 = (size_t)r0 * Nn + col0, i01 = (size_t)r0 * Nn + col1;
    size_t i10 = (size_t)r1 * Nn + col0, i11 = (size_t)r1 * Nn + col1;
    Out[i00] = w0 * (acc00[v] + bv0) + w1 * ctx[i00];
    Out[i01] = w0 * (acc01[v] + bv1) + w1 * ctx[i01];
    Out[i10] = w0 * (acc10[v] + bv0) + w1 * ctx[i10];
    Out[i11] = w0 * (acc11[v] + bv1) + w1 * ctx[i11];
  }
}

// ---------------------------------------------------------------------------
// Flash attention with cooperative async K/V staging.
// Block = 4 waves = one (b, h, 64-query group); wave w owns the w-th
// 16-query tile. Keys stepped by 32. Per key step:
//   K tile (32 keys x 64 ch) and V tile (64 ch x 32 keys, channel-major)
//   are async-copied to LDS once per block (double-buffered), then:
//   scores WMMA -> online softmax -> P relayout via per-wave LDS ->
//   P@V WMMA into 4 f32 accumulators (16x64 output per wave).
// ---------------------------------------------------------------------------
template <typename OutT>
__global__ void flash_attn_kernel(const bf16_t* __restrict__ Q,
                                  const bf16_t* __restrict__ Kmat,
                                  long kBatchStride, int ldk,
                                  const bf16_t* __restrict__ VT,
                                  long vtBatchStride, int ldvt,
                                  OutT* __restrict__ O,
                                  int numKeys, float scaleConst,
                                  const float* scalePtr) {
  __shared__ bf16_t ldsK[2][32][72];   // keys x channels, padded rows
  __shared__ bf16_t ldsV[2][64][40];   // channels x keys, padded rows
  __shared__ bf16_t Pbuf[4][16][40];   // per-wave 16x32 P tile, padded rows

  const int wave = threadIdx.x >> 5;
  const int lane = threadIdx.x & 31;
  const int tid  = threadIdx.x;

  const int grp = blockIdx.x;          // 0 .. B*H*(S/64)-1
  const int qg  = grp % (Sc / 64);
  const int bh  = grp / (Sc / 64);
  const int h   = bh % Hc;
  const int b   = bh / Hc;
  const int qt  = qg * 4 + wave;       // this wave's 16-query tile

  const float scl = scaleConst * (scalePtr ? scalePtr[0] : 1.0f);

  const bf16_t* Qb = Q + ((size_t)b * Sc + qt * 16) * Dc + h * HDc;
  const bf16_t* Kb = Kmat + (size_t)b * kBatchStride + h * HDc;
  const bf16_t* Vb = VT + (size_t)b * vtBatchStride + (size_t)h * HDc * ldvt;
  OutT* Ob = O + ((size_t)b * Sc + qt * 16) * Dc + h * HDc;

  const v16bf qa0 = load_frag(Qb, Dc, lane);        // channels 0..31
  const v16bf qa1 = load_frag(Qb + 32, Dc, lane);   // channels 32..63

  v8f acc[4];
  #pragma unroll
  for (int j = 0; j < 4; ++j) acc[j] = zero_v8f();
  float mrow[8], lrow[8];
  #pragma unroll
  for (int v = 0; v < 8; ++v) { mrow[v] = -3.0e38f; lrow[v] = 0.0f; }

  const int hf = lane >> 4, nn = lane & 15;

  // Cooperative async tile fill: 256 K-chunks + 256 V-chunks of 16B,
  // 4 async issues per thread per tile.
  auto issueTile = [&](int kb, int buf) {
    #pragma unroll
    for (int c = tid; c < 256; c += 128) {          // K: row = key, 8 chunks/row
      int row = c >> 3, col8 = c & 7;
      async_b128(Kb + (size_t)(kb + row) * ldk + col8 * 8,
                 &ldsK[buf][row][col8 * 8]);
    }
    #pragma unroll
    for (int c = tid; c < 256; c += 128) {          // V: row = channel, 4 chunks/row
      int ch = c >> 2, part = c & 3;
      async_b128(Vb + (size_t)ch * ldvt + kb + part * 8,
                 &ldsV[buf][ch][part * 8]);
    }
  };

  issueTile(0, 0);

  for (int kb = 0; kb < numKeys; kb += 32) {
    const int cur = (kb >> 5) & 1;
    if (kb + 32 < numKeys) {
      issueTile(kb + 32, cur ^ 1);                  // prefetch next tile
      asm volatile("s_wait_asynccnt 4" ::: "memory");  // tile `cur` landed
    } else {
      asm volatile("s_wait_asynccnt 0" ::: "memory");
    }
    __syncthreads();

    // ---- scores: S(16q x 32k) = Q(16x64) @ K(32x64)^T (from LDS) --------
    v16bf k00 = load_frag(&ldsK[cur][0][0],   72, lane);
    v16bf k01 = load_frag(&ldsK[cur][0][32],  72, lane);
    v16bf k10 = load_frag(&ldsK[cur][16][0],  72, lane);
    v16bf k11 = load_frag(&ldsK[cur][16][32], 72, lane);
    v8f s0 = wmma_bf16(qa0, k00, zero_v8f());
    s0 = wmma_bf16(qa1, k01, s0);
    v8f s1 = wmma_bf16(qa0, k10, zero_v8f());
    s1 = wmma_bf16(qa1, k11, s1);

    // ---- online softmax (row = query, striped across 16-lane halves) ----
    float p0[8], p1[8], corr[8];
    #pragma unroll
    for (int v = 0; v < 8; ++v) {
      float a = s0[v] * scl, c = s1[v] * scl;
      float mx = fmaxf(a, c);
      for (int mk = 8; mk >= 1; mk >>= 1)
        mx = fmaxf(mx, __shfl_xor(mx, mk, 32));
      float mnew = fmaxf(mrow[v], mx);
      corr[v] = __expf(mrow[v] - mnew);
      mrow[v] = mnew;
      p0[v] = __expf(a - mnew);
      p1[v] = __expf(c - mnew);
      float rs = p0[v] + p1[v];
      for (int mk = 8; mk >= 1; mk >>= 1)
        rs += __shfl_xor(rs, mk, 32);
      lrow[v] = lrow[v] * corr[v] + rs;
    }
    #pragma unroll
    for (int j = 0; j < 4; ++j)
      #pragma unroll
      for (int v = 0; v < 8; ++v) acc[j][v] *= corr[v];

    // ---- stage P (D-layout -> row-major 16x32 bf16 in LDS) --------------
    #pragma unroll
    for (int v = 0; v < 8; ++v) {
      int r = v + 8 * hf;
      Pbuf[wave][r][nn]      = (bf16_t)p0[v];
      Pbuf[wave][r][16 + nn] = (bf16_t)p1[v];
    }
    asm volatile("s_wait_dscnt 0" ::: "memory");    // wave-private LDS handoff

    // ---- O(16x64) += P(16x32) @ V(32x64)  (V from LDS, channel-major) ---
    v16bf pf = load_frag(&Pbuf[wave][0][0], 40, lane);
    #pragma unroll
    for (int j = 0; j < 4; ++j) {
      v16bf vf = load_frag(&ldsV[cur][j * 16][0], 40, lane);
      acc[j] = wmma_bf16(pf, vf, acc[j]);
    }
    __syncthreads();                                // release buffer `cur`
  }

  // ---- normalize + store ------------------------------------------------
  #pragma unroll
  for (int v = 0; v < 8; ++v) {
    float inv = 1.0f / lrow[v];
    int r = v + 8 * hf;
    #pragma unroll
    for (int j = 0; j < 4; ++j)
      Ob[(size_t)r * Dc + j * 16 + nn] = (OutT)(acc[j][v] * inv);
  }
}

// ---------------------------------------------------------------------------
// Host-side orchestration
// ---------------------------------------------------------------------------
extern "C" void kernel_launch(void* const* d_in, const int* in_sizes, int n_in,
                              void* d_out, int out_size, void* d_ws, size_t ws_size,
                              hipStream_t stream) {
  (void)in_sizes; (void)n_in; (void)out_size; (void)ws_size;

  const float* query  = (const float*)d_in[0];
  const float* key    = (const float*)d_in[1];
  const float* value  = (const float*)d_in[2];
  const float* vdata  = (const float*)d_in[3];
  const float* vk_w   = (const float*)d_in[4];
  const float* vk_bia = (const float*)d_in[5];
  const float* vv_w   = (const float*)d_in[6];
  const float* vv_bia = (const float*)d_in[7];
  const float* cscale = (const float*)d_in[8];
  const float* inw    = (const float*)d_in[9];
  const float* inb    = (const float*)d_in[10];
  const float* outw   = (const float*)d_in[11];
  const float* outb   = (const float*)d_in[12];
  float* out = (float*)d_out;

  // ---- carve workspace (~64 MB) ----
  char* wsp = (char*)d_ws;
  auto carveB = [&](size_t bytes) -> void* {
    void* r = (void*)wsp;
    wsp += (bytes + 255) & ~(size_t)255;
    return r;
  };
  const size_t BSD = (size_t)BSc * Dc;      // 1M
  const size_t NVD = (size_t)Nctx * VDc;    // 3M
  const size_t ND  = (size_t)Nctx * Dc;     // 4M

  bf16_t* qin  = (bf16_t*)carveB(BSD * 2);
  bf16_t* kin  = (bf16_t*)carveB(BSD * 2);
  bf16_t* vin  = (bf16_t*)carveB(BSD * 2);
  bf16_t* vdat = (bf16_t*)carveB(NVD * 2);
  bf16_t* winp = (bf16_t*)carveB((size_t)3 * Dc * Dc * 2);
  bf16_t* wvk  = (bf16_t*)carveB((size_t)Dc * VDc * 2);
  bf16_t* wvv  = (bf16_t*)carveB((size_t)Dc * VDc * 2);
  bf16_t* wout = (bf16_t*)carveB((size_t)Dc * Dc * 2);
  bf16_t* qpr  = (bf16_t*)carveB(BSD * 2);
  bf16_t* kpr  = (bf16_t*)carveB(BSD * 2);
  bf16_t* vpr  = (bf16_t*)carveB(BSD * 2);
  bf16_t* vkb  = (bf16_t*)carveB(ND * 2);
  bf16_t* vvb  = (bf16_t*)carveB(ND * 2);
  bf16_t* vtr  = (bf16_t*)carveB(BSD * 2);
  bf16_t* vvtr = (bf16_t*)carveB(ND * 2);
  bf16_t* attnb = (bf16_t*)carveB(BSD * 2);
  float*  ctxf  = (float*)carveB(BSD * 4);

  // ---- 1) fp32 -> bf16 conversions ----
  auto cvt = [&](const float* s, bf16_t* d, size_t n) {
    f32_to_bf16_kernel<<<(int)((n + 255) / 256), 256, 0, stream>>>(s, d, (int)n);
  };
  cvt(query, qin, BSD);
  cvt(key,   kin, BSD);
  cvt(value, vin, BSD);
  cvt(vdata, vdat, NVD);
  cvt(inw,   winp, (size_t)3 * Dc * Dc);
  cvt(vk_w,  wvk,  (size_t)Dc * VDc);
  cvt(vv_w,  wvv,  (size_t)Dc * VDc);
  cvt(outw,  wout, (size_t)Dc * Dc);

  dim3 blk(128);  // 4 waves per block

  // ---- 2) input projections q/k/v (M=1024, N=1024, K=1024) ----
  {
    int tiles = (BSc / 32) * (Dc / 32);  // 1024
    int grid = (tiles + 3) / 4;
    gemm_bt_bias_bf16out<<<grid, blk, 0, stream>>>(qin, winp, inb, qpr, BSc, Dc, Dc, tiles);
    gemm_bt_bias_bf16out<<<grid, blk, 0, stream>>>(kin, winp + (size_t)Dc * Dc, inb + Dc, kpr, BSc, Dc, Dc, tiles);
    gemm_bt_bias_bf16out<<<grid, blk, 0, stream>>>(vin, winp + (size_t)2 * Dc * Dc, inb + 2 * Dc, vpr, BSc, Dc, Dc, tiles);
  }

  // ---- 3) vector-bank projections vk/vv (M=4096, N=1024, K=768) ----
  {
    int tiles = (Nctx / 32) * (Dc / 32);  // 4096
    int grid = (tiles + 3) / 4;
    gemm_bt_bias_bf16out<<<grid, blk, 0, stream>>>(vdat, wvk, vk_bia, vkb, Nctx, Dc, VDc, tiles);
    gemm_bt_bias_bf16out<<<grid, blk, 0, stream>>>(vdat, wvv, vv_bia, vvb, Nctx, Dc, VDc, tiles);
  }

  // ---- 4) transposes for channel-major V operands ----
  transpose_v_kernel<<<(int)((BSD + 255) / 256), 256, 0, stream>>>(vpr, vtr);
  transpose_nm_kernel<<<(int)((ND + 255) / 256), 256, 0, stream>>>(vvb, vvtr);

  // ---- 5) flash attentions (block = (b,h,64-query group)) ----
  {
    int grid = Bc * Hc * (Sc / 64);  // 256 blocks, 4 waves each
    // base MHA: projected q/k, 512 keys per (b,h), scale 1/sqrt(HD)
    flash_attn_kernel<bf16_t><<<grid, blk, 0, stream>>>(
        qpr, kpr, (long)Sc * Dc, Dc,
        vtr, (long)Hc * HDc * Sc, Sc,
        attnb, Sc, 0.125f, (const float*)nullptr);
    // vector context: RAW query vs vk bank, 4096 keys, scale cscale/sqrt(HD)
    flash_attn_kernel<float><<<grid, blk, 0, stream>>>(
        qin, vkb, 0L, Dc,
        vvtr, 0L, Nctx,
        ctxf, Nctx, 0.125f, cscale);
  }

  // ---- 6) out-projection + combine ----
  {
    int tiles = (BSc / 32) * (Dc / 32);  // 1024
    int grid = (tiles + 3) / 4;
    gemm_final_combine<<<grid, blk, 0, stream>>>(attnb, wout, outb, ctxf, out,
                                                 BSc, Dc, Dc, tiles);
  }
}
